// NADE_42116449304657
// MI455X (gfx1250) — matrix operand order repaired and checked by reading, not compiled, not run on hardware
//
#include <hip/hip_runtime.h>
#include <hip/hip_bf16.h>
#include <math.h>

// ---------------------------------------------------------------------------
// NADE-style scan: B=1024, sequential D=1024, hidden H=256.
// Transcendental-bound (268M sigmoids). Sequential coupling is only the
// per-(b,h) prefix accumulator; the H-contraction runs on the matrix pipe
// (v_wmma_f32_16x16x32_bf16) and is split across 512 single-wave workgroups
// (64 b-tiles x 8 h-chunks) combined with global_atomic_add_f32.
// ---------------------------------------------------------------------------

typedef __attribute__((ext_vector_type(16))) __bf16 v16bf;
typedef __attribute__((ext_vector_type(8)))  float  v8f;
typedef __attribute__((ext_vector_type(8)))  int    v8i;

#define BB   1024   // batch
#define DD   1024   // visible dim (sequential)
#define HH   256    // hidden dim
#define BQ   16     // b-tile  (WMMA M)
#define BLK  16     // i-steps per block (WMMA N)
#define HCW  32     // h-chunk per workgroup (WMMA K, bf16 K=32)
#define NBLK (DD / BLK)

union BUnion { v8i i; v16bf h; };

// ============================ pass 0: zero scratch =========================
__global__ __launch_bounds__(256)
void nade_zero(float4* __restrict__ p, int n4)
{
    int i = blockIdx.x * 256 + threadIdx.x;
    if (i < n4) p[i] = make_float4(0.f, 0.f, 0.f, 0.f);
}

// ============== pass 1: recurrence + WMMA dot, 512 wave-workgroups =========
__global__ __launch_bounds__(32)
void nade_pass1(const float* __restrict__ x,        // [B, D]
                const float* __restrict__ W_h,      // [H, D]
                const float* __restrict__ b_h,      // [H]
                const float* __restrict__ alpha_w,  // [D, H]
                float* __restrict__ gDot)           // [B, D] scratch
{
    __shared__ float sWi[BLK][HCW];   // W slice, i-major: sWi[i_local][h_local]

    const int lane = threadIdx.x;        // 0..31 (single wave)
    const int grp  = lane >> 4;          // WMMA lane group
    const int bl   = lane & 15;          // A/C row (b) and C col (i) index
    const int bt   = blockIdx.x >> 3;    // b-tile      0..63
    const int hc   = blockIdx.x & 7;     // h-chunk     0..7
    const int b0   = bt * BQ;
    const int h0   = hc * HCW;

    // acc[t] laid out as the bf16 A-matrix (16x32):
    //   slot t -> K = ((t<8) ? t : t+8) + grp*8
    float acc[16];
    float bhv[16];
#pragma unroll
    for (int t = 0; t < 16; ++t) {
        acc[t] = 0.f;
        const int kk = ((t < 8) ? t : (t + 8)) + grp * 8;
        bhv[t] = b_h[h0 + kk];
    }

    for (int ib = 0; ib < NBLK; ++ib) {
        const int i0 = ib * BLK;
        __syncthreads();                 // S_NOP for single-wave WG

        // ---- stage W slice transposed to i-major in LDS ------------------
        {   // lane owns h-row (h0+lane): 16 contiguous i's, scatter to i-major
            const float4* src = (const float4*)(W_h + (size_t)(h0 + lane) * DD + i0);
            const float4 w0 = src[0], w1 = src[1], w2 = src[2], w3 = src[3];
            sWi[ 0][lane] = w0.x; sWi[ 1][lane] = w0.y; sWi[ 2][lane] = w0.z; sWi[ 3][lane] = w0.w;
            sWi[ 4][lane] = w1.x; sWi[ 5][lane] = w1.y; sWi[ 6][lane] = w1.z; sWi[ 7][lane] = w1.w;
            sWi[ 8][lane] = w2.x; sWi[ 9][lane] = w2.y; sWi[10][lane] = w2.z; sWi[11][lane] = w2.w;
            sWi[12][lane] = w3.x; sWi[13][lane] = w3.y; sWi[14][lane] = w3.z; sWi[15][lane] = w3.w;
        }

        // ---- lane-private register staging (no LDS needed) ---------------
        // x row b0+bl for the 16 steps of this block
        const float4* xs = (const float4*)(x + (size_t)(b0 + bl) * DD + i0);
        const float4 x0 = xs[0], x1 = xs[1], x2 = xs[2], x3 = xs[3];
        const float xrow[16] = { x0.x,x0.y,x0.z,x0.w, x1.x,x1.y,x1.z,x1.w,
                                 x2.x,x2.y,x2.z,x2.w, x3.x,x3.y,x3.z,x3.w };

        // alpha row i0+bl (this lane's one-hot column), packed bf16 once.
        // (bf16 32x16 B: slot s -> K = ((s<8) ? s : s+8) + grp*8, N=lane&15)
        BUnion bfull;
        {
            const float4* ar = (const float4*)(alpha_w + (size_t)(i0 + bl) * HH
                                               + h0 + grp * 8);
            const float4 a0 = ar[0], a1 = ar[1], a2 = ar[4], a3 = ar[5];
            const float av[16] = { a0.x,a0.y,a0.z,a0.w, a1.x,a1.y,a1.z,a1.w,
                                   a2.x,a2.y,a2.z,a2.w, a3.x,a3.y,a3.z,a3.w };
#pragma unroll
            for (int j = 0; j < 16; ++j) bfull.h[j] = (__bf16)av[j];
        }

        if (ib + 1 < NBLK) {             // global_prefetch_b8 of next block
            __builtin_prefetch(W_h + (size_t)(h0 + lane) * DD + i0 + BLK, 0, 1);
            __builtin_prefetch(alpha_w + (size_t)(i0 + BLK + bl) * HH + h0 + grp * 16, 0, 1);
            __builtin_prefetch(x + (size_t)(b0 + bl) * DD + i0 + BLK, 0, 1);
        }
        __syncthreads();

        // ---- 16 recurrence steps fused with WMMA H-contraction -----------
        v8f c = {};                      // dot[16b x 16i] tile (this h-chunk)
#pragma unroll
        for (int k = 0; k < BLK; ++k) {
            // hidden = sigmoid(exclusive prefix), packed to bf16 A tile
            v16bf a;
#pragma unroll
            for (int t = 0; t < 16; ++t) {
                const float e = __expf(-acc[t]);
                const float h = __builtin_amdgcn_rcpf(1.f + e);
                a[t] = (__bf16)h;
            }

            // one-hot column k: 8 x v_cndmask_b32 on the packed alpha row
            const bool sel = (bl == k);
            BUnion bm;
#pragma unroll
            for (int j2 = 0; j2 < 8; ++j2) bm.i[j2] = sel ? bfull.i[j2] : 0;

            c = __builtin_amdgcn_wmma_f32_16x16x32_bf16(
                    false, a, false, bm.h, (short)0, c, false, false);

            // recurrence: acc += x[b, i0+k] * W[h, i0+k] + b_h[h]
            const float  xv  = xrow[k];
            const float4* wr0 = (const float4*)&sWi[k][grp * 8];
            const float4* wr1 = (const float4*)&sWi[k][16 + grp * 8];
            const float4 q0 = wr0[0], q1 = wr0[1], q2 = wr1[0], q3 = wr1[1];
            const float wvv[16] = { q0.x,q0.y,q0.z,q0.w, q1.x,q1.y,q1.z,q1.w,
                                    q2.x,q2.y,q2.z,q2.w, q3.x,q3.y,q3.z,q3.w };
#pragma unroll
            for (int t = 0; t < 16; ++t)
                acc[t] += __fmaf_rn(xv, wvv[t], bhv[t]);
        }

        // ---- combine h-chunks in memory: global_atomic_add_f32 -----------
        // C layout: row M = vgpr + 8*grp (b), col N = lane&15 (i_local).
#pragma unroll
        for (int r = 0; r < 8; ++r)
            unsafeAtomicAdd(&gDot[(size_t)(b0 + r + grp * 8) * DD + i0 + bl], c[r]);
    }
}

// ================= pass 2: finalize sigmoid/log + row reduce ===============
__global__ __launch_bounds__(256)
void nade_pass2(const float* __restrict__ x,
                const float* __restrict__ alpha_b,
                const float* __restrict__ gDot,
                float* __restrict__ out)
{
    __shared__ float red[256];
    const int b = blockIdx.x, t = threadIdx.x;
    float ll = 0.f;
    for (int i = t; i < DD; i += 256) {
        const float d  = gDot[(size_t)b * DD + i] + alpha_b[i];
        const float e  = __expf(-d);
        const float p  = __builtin_amdgcn_rcpf(1.f + e);
        const float lp = __logf(p);
        const float xv = x[(size_t)b * DD + i];
        ll += xv * lp + (1.f - xv) * (1.f - lp);
    }
    red[t] = ll;
    __syncthreads();
#pragma unroll
    for (int s = 128; s > 0; s >>= 1) {
        if (t < s) red[t] += red[t + s];
        __syncthreads();
    }
    if (t == 0) out[b] = red[0];
}

// ============ fallback: fused single-pass version (no scratch) =============
__global__ __launch_bounds__(256)
void nade_fused(const float* __restrict__ x, const float* __restrict__ W_h,
                const float* __restrict__ b_h, const float* __restrict__ alpha_w,
                const float* __restrict__ alpha_b, float* __restrict__ out)
{
    __shared__ float sW[BLK][HH];
    __shared__ float sA[BLK][HH];
    __shared__ float sX[BQ][BLK];
    __shared__ float sDot[BQ][BLK];

    const int tid  = threadIdx.x;
    const int lane = tid & 31;
    const int wv   = tid >> 5;
    const int grp  = lane >> 4;
    const int bl   = lane & 15;
    const int b0   = blockIdx.x * BQ;
    const int h0   = wv * HCW;
    const int fb   = tid >> 4;
    const int fk   = tid & 15;

    float acc[16], bhv[16];
#pragma unroll
    for (int t = 0; t < 16; ++t) {
        acc[t] = 0.f;
        const int kk = ((t < 8) ? t : (t + 8)) + grp * 8;
        bhv[t] = b_h[h0 + kk];
    }
    sDot[fb][fk] = 0.f;
    float ll = 0.f;

    for (int ib = 0; ib < NBLK; ++ib) {
        const int i0 = ib * BLK;
        __syncthreads();
        {
            const float4* src = (const float4*)(W_h + (size_t)tid * DD + i0);
            const float4 w0 = src[0], w1 = src[1], w2 = src[2], w3 = src[3];
            sW[ 0][tid] = w0.x; sW[ 1][tid] = w0.y; sW[ 2][tid] = w0.z; sW[ 3][tid] = w0.w;
            sW[ 4][tid] = w1.x; sW[ 5][tid] = w1.y; sW[ 6][tid] = w1.z; sW[ 7][tid] = w1.w;
            sW[ 8][tid] = w2.x; sW[ 9][tid] = w2.y; sW[10][tid] = w2.z; sW[11][tid] = w2.w;
            sW[12][tid] = w3.x; sW[13][tid] = w3.y; sW[14][tid] = w3.z; sW[15][tid] = w3.w;
        }
        {
            const int il = tid >> 4, c0 = (tid & 15) * 16;
            const float4* src = (const float4*)(alpha_w + (size_t)(i0 + il) * HH + c0);
            float4* dst = (float4*)&sA[il][c0];
            dst[0] = src[0]; dst[1] = src[1]; dst[2] = src[2]; dst[3] = src[3];
        }
        sX[fb][fk] = x[(size_t)(b0 + fb) * DD + i0 + fk];
        __syncthreads();

        v8f c = {};
        for (int k = 0; k < BLK; ++k) {
            v16bf a;
#pragma unroll
            for (int t = 0; t < 16; ++t) {
                const float e = __expf(-acc[t]);
                const float h = __builtin_amdgcn_rcpf(1.f + e);
                a[t] = (__bf16)h;
            }
            const float4* ar = (const float4*)&sA[k][h0 + grp * 8];
            const float4 p0 = ar[0], p1 = ar[1], p2 = ar[4], p3 = ar[5];
            const float av[16] = { p0.x,p0.y,p0.z,p0.w, p1.x,p1.y,p1.z,p1.w,
                                   p2.x,p2.y,p2.z,p2.w, p3.x,p3.y,p3.z,p3.w };
            const bool sel = (bl == k);
            v16bf bm;
#pragma unroll
            for (int j = 0; j < 16; ++j)
                bm[j] = sel ? (__bf16)av[j] : (__bf16)0.f;
            c = __builtin_amdgcn_wmma_f32_16x16x32_bf16(
                    false, a, false, bm, (short)0, c, false, false);
            const float  xv  = sX[bl][k];
            const float4* wr0 = (const float4*)&sW[k][h0 + grp * 8];
            const float4* wr1 = (const float4*)&sW[k][h0 + 16 + grp * 8];
            const float4 q0 = wr0[0], q1 = wr0[1], q2 = wr1[0], q3 = wr1[1];
            const float wvv[16] = { q0.x,q0.y,q0.z,q0.w, q1.x,q1.y,q1.z,q1.w,
                                    q2.x,q2.y,q2.z,q2.w, q3.x,q3.y,q3.z,q3.w };
#pragma unroll
            for (int t = 0; t < 16; ++t)
                acc[t] += __fmaf_rn(xv, wvv[t], bhv[t]);
        }
#pragma unroll
        for (int r = 0; r < 8; ++r)
            atomicAdd(&sDot[r + grp * 8][bl], c[r]);
        __syncthreads();
        {
            const float d  = sDot[fb][fk] + alpha_b[i0 + fk];
            const float e  = __expf(-d);
            const float p  = __builtin_amdgcn_rcpf(1.f + e);
            const float lp = __logf(p);
            const float xq = sX[fb][fk];
            ll += xq * lp + (1.f - xq) * (1.f - lp);
            sDot[fb][fk] = 0.f;
        }
    }
    __syncthreads();
    sDot[fb][fk] = ll;
    __syncthreads();
    if (tid < BQ) {
        float s = 0.f;
#pragma unroll
        for (int kk = 0; kk < BLK; ++kk) s += sDot[tid][kk];
        out[b0 + tid] = s;
    }
}

// ================================ launcher =================================
extern "C" void kernel_launch(void* const* d_in, const int* in_sizes, int n_in,
                              void* d_out, int out_size, void* d_ws, size_t ws_size,
                              hipStream_t stream) {
    const float* x   = (const float*)d_in[0];  // [B, D]
    const float* W_h = (const float*)d_in[1];  // [H, D]
    const float* b_h = (const float*)d_in[2];  // [H]
    const float* aw  = (const float*)d_in[3];  // [D, H]
    const float* ab  = (const float*)d_in[4];  // [D]
    float* out = (float*)d_out;                // [B]

    const size_t need = (size_t)BB * DD * sizeof(float);   // 4 MB dot scratch
    if (d_ws != nullptr && ws_size >= need) {
        float* gDot = (float*)d_ws;
        const int n4 = (BB * DD) / 4;
        hipLaunchKernelGGL(nade_zero, dim3((n4 + 255) / 256), dim3(256), 0, stream,
                           (float4*)gDot, n4);
        hipLaunchKernelGGL(nade_pass1, dim3((BB / BQ) * (HH / HCW)), dim3(32), 0,
                           stream, x, W_h, b_h, aw, gDot);
        hipLaunchKernelGGL(nade_pass2, dim3(BB), dim3(256), 0, stream,
                           x, ab, gDot, out);
    } else {
        hipLaunchKernelGGL(nade_fused, dim3(BB / BQ), dim3(256), 0, stream,
                           x, W_h, b_h, aw, ab, out);
    }
}
